// Block_69320772157513
// MI455X (gfx1250) — matrix operand order repaired and checked by reading, not compiled, not run on hardware
//
#include <hip/hip_runtime.h>
#include <math.h>

typedef __attribute__((ext_vector_type(16))) __bf16 v16bf;
typedef __attribute__((ext_vector_type(8)))  float  v8f;
typedef __attribute__((ext_vector_type(4)))  unsigned int u32x4;
typedef unsigned short u16;

// ---------------- helpers ----------------
__device__ __forceinline__ u16 f2bf(float f) {
    unsigned int u = __builtin_bit_cast(unsigned int, f);
    unsigned int r = u + 0x7FFFu + ((u >> 16) & 1u);   // round-to-nearest-even
    return (u16)(r >> 16);
}

__device__ __forceinline__ v16bf ld_frag(const u16* p, int step_halves) {
    union { u32x4 u[2]; v16bf v; } r;
    r.u[0] = *(const u32x4*)(p);
    r.u[1] = *(const u32x4*)(p + step_halves);
    return r.v;
}

__device__ __forceinline__ v8f wmma_bf16(v16bf a, v16bf b, v8f c) {
    return __builtin_amdgcn_wmma_f32_16x16x32_bf16(false, a, false, b, (short)0, c, false, false);
}

// CDNA5 async global->LDS copy (ASYNCcnt-tracked, no VGPR staging).
// LDS dest: low 32 bits of generic shared pointer == LDS byte offset.
__device__ __forceinline__ void async_b128(unsigned lds_addr, const u16* gaddr) {
    asm volatile("global_load_async_to_lds_b128 %0, %1, off"
                 :: "v"(lds_addr), "v"(gaddr) : "memory");
}
__device__ __forceinline__ void wait_async0() {
    asm volatile("s_wait_asynccnt 0" ::: "memory");
}

// ---------------- generic batched GEMM: C = alpha * A[M,K] * Bt[N,K]^T (+bias +residual, opt GELU)
// A: bf16 row-major, lda; Bt: bf16 [N][K] row-major, ldb; out f32 and/or bf16, ldc.
// Block tile 64x256, 8 waves, each wave 32x64 (8 WMMA per 32-K step).
// Double-buffered LDS filled by async loads: one barrier per K-step.
#define TILE_M 64
#define TILE_N 256
#define BK     32
#define LDSS   40                      // padded LDS row stride (80B: aligned, conflict-free)
#define A_ELEMS   (TILE_M * LDSS)      // 2560 halves
#define B_ELEMS   (TILE_N * LDSS)      // 10240 halves
#define BUF_ELEMS (A_ELEMS + B_ELEMS)  // 12800 halves = 25600 B per buffer

__global__ __launch_bounds__(256)
void gemm_bf16_kernel(const u16* __restrict__ A, const u16* __restrict__ Bt,
                      const float* __restrict__ bias, const float* __restrict__ residual,
                      float* __restrict__ outF, u16* __restrict__ outH,
                      int M, int N, int K, int lda, int ldb, int ldc,
                      int Hh, long long aBS, long long aHS, long long bBS, long long bHS,
                      long long cBS, long long cHS, float alpha, int doGelu)
{
    __shared__ u16 lds[2 * BUF_ELEMS];

    const int tid   = threadIdx.x;
    const int lane  = tid & 31;
    const int wave  = tid >> 5;
    const int waveM = wave >> 2;   // 0..1 -> 32-row slab
    const int waveN = wave & 3;    // 0..3 -> 64-col slab

    const int z  = blockIdx.z;
    const int bb = z / Hh, hh = z % Hh;
    const u16* Ab = A  + (long long)bb * aBS + (long long)hh * aHS;
    const u16* Bb = Bt + (long long)bb * bBS + (long long)hh * bHS;
    const long long coff = (long long)bb * cBS + (long long)hh * cHS;

    const int m0 = blockIdx.y * TILE_M;
    const int n0 = blockIdx.x * TILE_N;

    // global->LDS tile mapping: thread t loads row t>>2, 8-half chunk (t&3)*8
    const int grow = tid >> 2;           // 0..63
    const int gch  = (tid & 3) * 8;      // 0,8,16,24 halves
    const u16* Agp  = Ab + (long long)(m0 + grow) * lda + gch;
    const u16* Bgp  = Bb + (long long)(n0 + grow) * ldb + gch;
    const long long bRow64 = (long long)64 * ldb;

    const unsigned ldsBase = (unsigned)(uintptr_t)(&lds[0]);
    const unsigned aDst = ldsBase + (unsigned)(grow * LDSS + gch) * 2u;
    const unsigned bDst = ldsBase + (unsigned)(A_ELEMS + grow * LDSS + gch) * 2u;

    auto issue = [&](int buf, int k0) {
        const unsigned boff = (unsigned)buf * (BUF_ELEMS * 2u);
        async_b128(aDst + boff,                      Agp + k0);
        async_b128(bDst + boff,                      Bgp + k0);
        async_b128(bDst + boff +  64u * LDSS * 2u,   Bgp + bRow64     + k0);
        async_b128(bDst + boff + 128u * LDSS * 2u,   Bgp + 2 * bRow64 + k0);
        async_b128(bDst + boff + 192u * LDSS * 2u,   Bgp + 3 * bRow64 + k0);
    };

    // fragment base pointers (ISA 16-bit A/B layouts)
    const int r15     = lane & 15;
    const int halfSel = (lane < 16) ? 0 : 1;
    const u16* aF = &lds[(waveM * 32 + r15) * LDSS + halfSel * 8];           // chunks +0,+16
    const u16* bF = &lds[A_ELEMS + (waveN * 64 + r15) * LDSS + halfSel * 8];  // chunks +0,+16

    v8f acc[2][4];
#pragma unroll
    for (int mi = 0; mi < 2; ++mi)
#pragma unroll
        for (int ni = 0; ni < 4; ++ni) acc[mi][ni] = (v8f){};

    auto compute = [&](int buf) {
        const u16* a = aF + buf * BUF_ELEMS;
        const u16* b = bF + buf * BUF_ELEMS;
        v16bf a0 = ld_frag(a, 16);
        v16bf a1 = ld_frag(a + 16 * LDSS, 16);
#pragma unroll
        for (int ni = 0; ni < 4; ++ni) {
            v16bf bn = ld_frag(b + ni * 16 * LDSS, 16);
            acc[0][ni] = wmma_bf16(a0, bn, acc[0][ni]);
            acc[1][ni] = wmma_bf16(a1, bn, acc[1][ni]);
        }
    };

    // prologue
    issue(0, 0);
    wait_async0();
    __syncthreads();

    int k0 = 0;
    for (;;) {
        if (k0 + BK < K) issue(1, k0 + BK);
        compute(0);
        wait_async0();
        __syncthreads();
        k0 += BK;
        if (k0 >= K) break;

        if (k0 + BK < K) issue(0, k0 + BK);
        compute(1);
        wait_async0();
        __syncthreads();
        k0 += BK;
        if (k0 >= K) break;
    }

    // epilogue: C layout — VGPR r: M = r (lanes 0-15) / r+8 (lanes 16-31); N = lane&15
    const int nbase = n0 + waveN * 64 + r15;
    const int mbase = m0 + waveM * 32 + halfSel * 8;
#pragma unroll
    for (int mi = 0; mi < 2; ++mi) {
#pragma unroll
        for (int ni = 0; ni < 4; ++ni) {
            const int n = nbase + ni * 16;
            const float bv = bias ? bias[n] : 0.0f;
#pragma unroll
            for (int r = 0; r < 8; ++r) {
                const int m = mbase + mi * 16 + r;
                const long long off = coff + (long long)m * ldc + n;
                float val = acc[mi][ni][r] * alpha + bv;
                if (residual) val += residual[off];
                if (doGelu)   val = 0.5f * val * (1.0f + erff(val * 0.70710678118654752f));
                if (outF) outF[off] = val;
                if (outH) outH[off] = f2bf(val);
            }
        }
    }
}

// ---------------- layernorm: f32 row -> bf16 row ----------------
__global__ __launch_bounds__(256)
void ln_kernel(const float* __restrict__ x, const float* __restrict__ w,
               const float* __restrict__ b, u16* __restrict__ out, int Cd)
{
    __shared__ float r1[256], r2[256];
    const int tid = threadIdx.x;
    const long long row = blockIdx.x;
    const float* xr = x + row * (long long)Cd;
    u16* o = out + row * (long long)Cd;

    float s = 0.f, s2 = 0.f;
    float xv[8];
    const int per = Cd / 256;  // 8 for C=2048
#pragma unroll
    for (int i = 0; i < 8; ++i) {
        float v = (i < per) ? xr[tid + i * 256] : 0.f;
        xv[i] = v; s += v; s2 += v * v;
    }
    r1[tid] = s; r2[tid] = s2; __syncthreads();
    for (int st = 128; st > 0; st >>= 1) {
        if (tid < st) { r1[tid] += r1[tid + st]; r2[tid] += r2[tid + st]; }
        __syncthreads();
    }
    const float mu   = r1[0] / (float)Cd;
    const float var  = r2[0] / (float)Cd - mu * mu;
    const float rstd = rsqrtf(var + 1e-5f);
#pragma unroll
    for (int i = 0; i < 8; ++i) {
        if (i < per) {
            int j = tid + i * 256;
            o[j] = f2bf((xv[i] - mu) * rstd * w[j] + b[j]);
        }
    }
}

// ---------------- masked softmax with all-masked->0, f32 scores -> bf16 probs ----------------
__global__ __launch_bounds__(256)
void softmax_kernel(const float* __restrict__ S, const int* __restrict__ mask,
                    u16* __restrict__ P, int T, int Hh)
{
    __shared__ float red[256];
    const int tid = threadIdx.x;
    const long long row = blockIdx.x;           // (b*H + h)*T + q
    const long long q   = row % T;
    const long long bh  = row / T;
    const long long b   = bh / Hh;
    const float* s   = S + row * (long long)T;
    const int* mrow  = mask + (b * (long long)T + q) * T;
    u16* p = P + row * (long long)T;

    float lmax = -INFINITY;
    for (int j = tid; j < T; j += 256)
        if (mrow[j] != 0) lmax = fmaxf(lmax, s[j]);
    red[tid] = lmax; __syncthreads();
    for (int st = 128; st > 0; st >>= 1) {
        if (tid < st) red[tid] = fmaxf(red[tid], red[tid + st]);
        __syncthreads();
    }
    const float m = red[0]; __syncthreads();
    if (m == -INFINITY) {                        // fully masked row -> zeros
        for (int j = tid; j < T; j += 256) p[j] = 0;
        return;
    }
    float lsum = 0.f;
    for (int j = tid; j < T; j += 256)
        if (mrow[j] != 0) lsum += expf(s[j] - m);
    red[tid] = lsum; __syncthreads();
    for (int st = 128; st > 0; st >>= 1) {
        if (tid < st) red[tid] += red[tid + st];
        __syncthreads();
    }
    const float inv = 1.0f / red[0];
    for (int j = tid; j < T; j += 256) {
        float v = (mrow[j] != 0) ? expf(s[j] - m) * inv : 0.0f;
        p[j] = f2bf(v);
    }
}

// ---------------- weight transpose-convert: f32 [K,N] -> bf16 [N,K] ----------------
__global__ __launch_bounds__(256)
void wtrans_kernel(const float* __restrict__ W, u16* __restrict__ Wt, int Kd, int Nd)
{
    __shared__ u16 t[32][33];
    const int tx = threadIdx.x & 31, ty = threadIdx.x >> 5;
    const int n0 = blockIdx.x * 32, k0 = blockIdx.y * 32;
#pragma unroll
    for (int i = 0; i < 4; ++i)
        t[tx][ty + 8 * i] = f2bf(W[(long long)(k0 + ty + 8 * i) * Nd + n0 + tx]);
    __syncthreads();
#pragma unroll
    for (int i = 0; i < 4; ++i)
        Wt[(long long)(n0 + ty + 8 * i) * Kd + k0 + tx] = t[ty + 8 * i][tx];
}

// ---------------- per-head transpose: v[B,T,H,DH] (bf16) -> vt[B,H,DH,T] (bf16) ----------------
__global__ __launch_bounds__(256)
void vtrans_kernel(const u16* __restrict__ V, u16* __restrict__ Vt,
                   int T, int C, int DH, int Hh)
{
    const int z = blockIdx.z;
    const int b = z / Hh, h = z % Hh;
    const u16* src = V + (long long)b * T * C + (long long)h * DH;
    u16* dst = Vt + (long long)z * DH * T;
    __shared__ u16 t[32][33];
    const int tx = threadIdx.x & 31, ty = threadIdx.x >> 5;
    const int t0 = blockIdx.x * 32, d0 = blockIdx.y * 32;
#pragma unroll
    for (int i = 0; i < 4; ++i)
        t[tx][ty + 8 * i] = src[(long long)(t0 + ty + 8 * i) * C + d0 + tx];
    __syncthreads();
#pragma unroll
    for (int i = 0; i < 4; ++i)
        dst[(long long)(d0 + ty + 8 * i) * T + t0 + tx] = t[ty + 8 * i][tx];
}

// ---------------- host launcher ----------------
extern "C" void kernel_launch(void* const* d_in, const int* in_sizes, int n_in,
                              void* d_out, int out_size, void* d_ws, size_t ws_size,
                              hipStream_t stream)
{
    const float* x    = (const float*)d_in[0];
    const int*   mask = (const int*)  d_in[1];
    const float* Wq   = (const float*)d_in[2];  const float* bq = (const float*)d_in[3];
    const float* Wk   = (const float*)d_in[4];  const float* bk = (const float*)d_in[5];
    const float* Wv   = (const float*)d_in[6];  const float* bv = (const float*)d_in[7];
    const float* Wo   = (const float*)d_in[8];  const float* bo = (const float*)d_in[9];
    const float* ln1w = (const float*)d_in[10]; const float* ln1b = (const float*)d_in[11];
    const float* ln2w = (const float*)d_in[12]; const float* ln2b = (const float*)d_in[13];
    const float* W1   = (const float*)d_in[14]; const float* b1 = (const float*)d_in[15];
    const float* W2   = (const float*)d_in[16]; const float* b2 = (const float*)d_in[17];
    float* out = (float*)d_out;

    constexpr int Bb = 4, T = 1024, C = 2048, Hh = 4, DH = 512;
    const int M = Bb * T;                       // 4096

    char* ws = (char*)d_ws;
    size_t off = 0;
    auto alloc = [&](size_t bytes) -> char* {
        char* p = ws + off;
        off = (off + bytes + 255) & ~(size_t)255;
        return p;
    };
    u16*   wq_t = (u16*)alloc((size_t)C * C * 2);
    u16*   wk_t = (u16*)alloc((size_t)C * C * 2);
    u16*   wv_t = (u16*)alloc((size_t)C * C * 2);
    u16*   wo_t = (u16*)alloc((size_t)C * C * 2);
    u16*   w1_t = (u16*)alloc((size_t)C * 4 * C * 2);
    u16*   w2_t = (u16*)alloc((size_t)C * 4 * C * 2);
    u16*   xn   = (u16*)alloc((size_t)M * C * 2);
    u16*   qb   = (u16*)alloc((size_t)M * C * 2);
    u16*   kb   = (u16*)alloc((size_t)M * C * 2);
    u16*   vb   = (u16*)alloc((size_t)M * C * 2);
    u16*   vt   = (u16*)alloc((size_t)M * C * 2);
    float* Sf   = (float*)alloc((size_t)Bb * Hh * T * T * 4);
    u16*   Pb   = (u16*)alloc((size_t)Bb * Hh * T * T * 2);
    u16*   yb   = (u16*)alloc((size_t)M * C * 2);
    float* x2f  = (float*)alloc((size_t)M * C * 4);
    u16*   xn2  = (u16*)alloc((size_t)M * C * 2);
    u16*   hb   = (u16*)alloc((size_t)M * 4 * C * 2);

    const dim3 blk(256);

    // weights -> bf16 transposed [N][K]
    wtrans_kernel<<<dim3(C/32,     C/32,     1), blk, 0, stream>>>(Wq, wq_t, C, C);
    wtrans_kernel<<<dim3(C/32,     C/32,     1), blk, 0, stream>>>(Wk, wk_t, C, C);
    wtrans_kernel<<<dim3(C/32,     C/32,     1), blk, 0, stream>>>(Wv, wv_t, C, C);
    wtrans_kernel<<<dim3(C/32,     C/32,     1), blk, 0, stream>>>(Wo, wo_t, C, C);
    wtrans_kernel<<<dim3(4*C/32,   C/32,     1), blk, 0, stream>>>(W1, w1_t, C, 4*C);
    wtrans_kernel<<<dim3(C/32,     4*C/32,   1), blk, 0, stream>>>(W2, w2_t, 4*C, C);

    // LN1
    ln_kernel<<<M, blk, 0, stream>>>(x, ln1w, ln1b, xn, C);

    // q,k,v projections (bf16 outputs)
    gemm_bf16_kernel<<<dim3(C/TILE_N, M/TILE_M, 1), blk, 0, stream>>>(
        xn, wq_t, bq, nullptr, nullptr, qb, M, C, C, C, C, C,
        1, 0, 0, 0, 0, 0, 0, 1.0f, 0);
    gemm_bf16_kernel<<<dim3(C/TILE_N, M/TILE_M, 1), blk, 0, stream>>>(
        xn, wk_t, bk, nullptr, nullptr, kb, M, C, C, C, C, C,
        1, 0, 0, 0, 0, 0, 0, 1.0f, 0);
    gemm_bf16_kernel<<<dim3(C/TILE_N, M/TILE_M, 1), blk, 0, stream>>>(
        xn, wv_t, bv, nullptr, nullptr, vb, M, C, C, C, C, C,
        1, 0, 0, 0, 0, 0, 0, 1.0f, 0);

    // v -> per-head transposed [B,H,DH,T]
    vtrans_kernel<<<dim3(T/32, DH/32, Bb*Hh), blk, 0, stream>>>(vb, vt, T, C, DH, Hh);

    // S = q @ k^T * (1/sqrt(DH))   (batched over B*H; A/B are head-strided views)
    gemm_bf16_kernel<<<dim3(T/TILE_N, T/TILE_M, Bb*Hh), blk, 0, stream>>>(
        qb, kb, nullptr, nullptr, Sf, nullptr, T, T, DH, C, C, T,
        Hh, (long long)T*C, DH, (long long)T*C, DH,
        (long long)Hh*T*T, (long long)T*T, 0.044194173824159216f, 0);

    // masked softmax -> bf16 P
    softmax_kernel<<<Bb*Hh*T, blk, 0, stream>>>(Sf, mask, Pb, T, Hh);

    // y = P @ V  (written strided into [B,T,H,DH] == [B,T,C])
    gemm_bf16_kernel<<<dim3(DH/TILE_N, T/TILE_M, Bb*Hh), blk, 0, stream>>>(
        Pb, vt, nullptr, nullptr, nullptr, yb, T, DH, T, T, T, C,
        Hh, (long long)Hh*T*T, (long long)T*T,
        (long long)Hh*DH*T, (long long)DH*T,
        (long long)T*C, DH, 1.0f, 0);

    // x2 = x + y @ Wo + bo   (f32)
    gemm_bf16_kernel<<<dim3(C/TILE_N, M/TILE_M, 1), blk, 0, stream>>>(
        yb, wo_t, bo, x, x2f, nullptr, M, C, C, C, C, C,
        1, 0, 0, 0, 0, 0, 0, 1.0f, 0);

    // LN2
    ln_kernel<<<M, blk, 0, stream>>>(x2f, ln2w, ln2b, xn2, C);

    // h = gelu(xn2 @ W1 + b1)  (bf16)
    gemm_bf16_kernel<<<dim3(4*C/TILE_N, M/TILE_M, 1), blk, 0, stream>>>(
        xn2, w1_t, b1, nullptr, nullptr, hb, M, 4*C, C, C, C, 4*C,
        1, 0, 0, 0, 0, 0, 0, 1.0f, 1);

    // out = x2 + h @ W2 + b2  (f32 -> d_out)
    gemm_bf16_kernel<<<dim3(C/TILE_N, M/TILE_M, 1), blk, 0, stream>>>(
        hb, w2_t, b2, x2f, out, nullptr, M, C, 4*C, 4*C, 4*C, C,
        1, 0, 0, 0, 0, 0, 0, 1.0f, 0);
}
